// PointNetSAModule_6176162972233
// MI455X (gfx1250) — matrix operand order repaired and checked by reading, not compiled, not run on hardware
//
#include <hip/hip_runtime.h>
#include <hip/hip_bf16.h>

// MI455X / gfx1250, wave32. WMMA via __builtin_amdgcn_wmma_f32_16x16x32_bf16.

typedef __attribute__((ext_vector_type(16))) __bf16 bf16x16;
typedef __attribute__((ext_vector_type(8)))  float  f32x8;

#define IN_DIMC  128
#define OUT_DIMC 256
#define KNBR     16
#define RADIUS_F 0.1f
#define BATCH    4
#define NPTS     8192
#define MPTS     2048     // NPTS / 4
#define TILE_N   64

// ---------------------------------------------------------------------------
// Kernel 1: furthest point sampling. One block per batch, points in registers.
// ---------------------------------------------------------------------------
__global__ __launch_bounds__(1024) void fps_kernel(const float* __restrict__ xyz,
                                                   float* __restrict__ child_xyz,
                                                   int*   __restrict__ child_idx)
{
    const int b = blockIdx.x;
    const int t = threadIdx.x;
    const int PT = NPTS / 1024;          // 8 points per thread
    const float* xb = xyz + (size_t)b * NPTS * 3;

    float px[PT], py[PT], pz[PT], md[PT];
#pragma unroll
    for (int j = 0; j < PT; ++j) {
        int n = j * 1024 + t;
        px[j] = xb[n*3+0]; py[j] = xb[n*3+1]; pz[j] = xb[n*3+2];
        md[j] = 1e10f;
    }

    __shared__ float s_bv[32];
    __shared__ int   s_bi[32];
    __shared__ int   s_far;
    if (t == 0) s_far = 0;
    __syncthreads();

    for (int i = 0; i < MPTS; ++i) {
        int far = s_far;
        if (t == 0) {
            child_idx[b*MPTS + i] = far;
            child_xyz[((size_t)b*MPTS + i)*3 + 0] = xb[far*3+0];
            child_xyz[((size_t)b*MPTS + i)*3 + 1] = xb[far*3+1];
            child_xyz[((size_t)b*MPTS + i)*3 + 2] = xb[far*3+2];
        }
        float cx = xb[far*3+0], cy = xb[far*3+1], cz = xb[far*3+2];

        float bv = -1.0f; int bi = 0x7fffffff;
#pragma unroll
        for (int j = 0; j < PT; ++j) {
            float dx = px[j]-cx, dy = py[j]-cy, dz = pz[j]-cz;
            float d  = dx*dx + dy*dy + dz*dz;
            md[j] = fminf(md[j], d);
            if (md[j] > bv) { bv = md[j]; bi = j*1024 + t; }
        }
        // wave32 argmax (lowest index wins ties, matching jnp.argmax)
#pragma unroll
        for (int off = 16; off > 0; off >>= 1) {
            float ov = __shfl_down(bv, off, 32);
            int   oi = __shfl_down(bi, off, 32);
            if (ov > bv || (ov == bv && oi < bi)) { bv = ov; bi = oi; }
        }
        if ((t & 31) == 0) { s_bv[t >> 5] = bv; s_bi[t >> 5] = bi; }
        __syncthreads();
        if (t < 32) {
            bv = s_bv[t]; bi = s_bi[t];
#pragma unroll
            for (int off = 16; off > 0; off >>= 1) {
                float ov = __shfl_down(bv, off, 32);
                int   oi = __shfl_down(bi, off, 32);
                if (ov > bv || (ov == bv && oi < bi)) { bv = ov; bi = oi; }
            }
            if (t == 0) s_far = bi;
        }
        __syncthreads();
    }
}

// ---------------------------------------------------------------------------
// Weight pre-pack: f32 row-major -> bf16 in A-fragment lane order, so the GEMM
// fetches each A fragment as 32 contiguous bytes per lane (2x global_load_b128,
// fully coalesced across the wave, zero VALU shuffling in the hot loop).
// Layout: [mt][kc][lane][i], i -> k = 16*(lane/16) + i (contiguous k order, same as the B LDS layout).
// ---------------------------------------------------------------------------
__global__ __launch_bounds__(256) void pack_w_kernel(const float* __restrict__ W,
                                                     __bf16* __restrict__ out,
                                                     int ld /* K total */)
{
    int tid = blockIdx.x * 256 + threadIdx.x;
    int kcn = ld / 32;                               // k-chunks
    int total = 16 * kcn * 32 * 16;                  // mt * kc * lane * i
    if (tid >= total) return;
    int i    = tid & 15;
    int lane = (tid >> 4) & 31;
    int kc   = (tid >> 9) & (kcn - 1);               // kcn is 4 or 8
    int mt   = tid / (kcn << 9);
    int row   = mt * 16 + (lane & 15);
    int kbase = (lane < 16) ? 0 : 16;
    int k = kc * 32 + kbase + i;
    out[tid] = (__bf16)W[(size_t)row * ld + k];
}

// ---------------------------------------------------------------------------
// Kernel 2: fused two-layer 1x1-conv MLP, bf16 WMMA with f32 accumulate.
// Grid: B*(N/64) blocks, 256 threads (8 waves). Wave w owns channel tiles
// mt = {w, w+8}. LDS buffers (X bf16, Y1 bf16) are kept in B-fragment-swizzled
// layout [(k>>4)][n][k&15] so each lane's 16 B elements are 32 contiguous
// bytes -> 2x ds_load_b128 per fragment, no element moves.
// Output y2 stored (B, N, 256) so grouping gathers are coalesced.
// ---------------------------------------------------------------------------
__device__ inline bf16x16 load_frag32B(const __bf16* __restrict__ p)
{
    union { uint4 u[2]; bf16x16 v; } c;
    c.u[0] = *(const uint4*)(p);
    c.u[1] = *(const uint4*)(p + 8);
    return c.v;
}

__global__ __launch_bounds__(256) void mlp_kernel(const float* __restrict__ feats,
                                                  const __bf16* __restrict__ w1f,
                                                  const float* __restrict__ b1,
                                                  const __bf16* __restrict__ w2f,
                                                  const float* __restrict__ b2,
                                                  float* __restrict__ y2)
{
    __shared__ __align__(16) __bf16 s_mem[16384];   // 32 KB, reused 3x
    __bf16* s_xb = s_mem;               // X  tile: kgrp 0..7  -> 16 KB
    __bf16* s_y1 = s_mem;               // Y1 tile: kgrp 0..15 -> 32 KB
    float*  s_o  = (float*)s_mem;       // out transpose: 32 x 256 f32 = 32 KB

    const int bi   = blockIdx.x >> 7;          // batch
    const int p0   = (blockIdx.x & 127) * TILE_N;
    const int lane = threadIdx.x & 31;
    const int wave = threadIdx.x >> 5;         // 0..7
    const int nl   = lane & 15;
    const int hi8  = (lane < 16) ? 0 : 8;
    const int hlf  = (lane < 16) ? 0 : 1;

    // --- stage X tile: coalesced f32 reads, convert once, swizzled bf16 LDS ---
    for (int idx = threadIdx.x; idx < IN_DIMC * TILE_N / 4; idx += 256) {
        int k  = idx >> 4;                 // 0..127
        int n4 = (idx & 15) * 4;           // 0..60
        float4 f = *(const float4*)(feats + ((size_t)bi * IN_DIMC + k) * NPTS + p0 + n4);
        int base = (k >> 4) * 1024 + (k & 15);
        s_xb[base + (n4+0)*16] = (__bf16)f.x;
        s_xb[base + (n4+1)*16] = (__bf16)f.y;
        s_xb[base + (n4+2)*16] = (__bf16)f.z;
        s_xb[base + (n4+3)*16] = (__bf16)f.w;
    }
    __syncthreads();

    // --- layer 1: Y1 = relu(W1 * X + b1), K = 128 ---
    f32x8 acc[2][4] = {};
#pragma unroll
    for (int kc = 0; kc < IN_DIMC / 32; ++kc) {
        bf16x16 a0 = load_frag32B(w1f + (((wave    ) * 4 + kc) * 32 + lane) * 16);
        bf16x16 a1 = load_frag32B(w1f + (((wave + 8) * 4 + kc) * 32 + lane) * 16);
#pragma unroll
        for (int nt = 0; nt < 4; ++nt) {
            bf16x16 bb = load_frag32B(s_xb + (kc*2 + hlf) * 1024 + (nt*16 + nl) * 16);
            acc[0][nt] = __builtin_amdgcn_wmma_f32_16x16x32_bf16(
                false, a0, false, bb, (short)0, acc[0][nt], false, false);
            acc[1][nt] = __builtin_amdgcn_wmma_f32_16x16x32_bf16(
                false, a1, false, bb, (short)0, acc[1][nt], false, false);
        }
    }
    __syncthreads();   // everyone done reading s_xb

    // --- relu(Y1)+bias -> swizzled bf16 LDS; one 16B ds_store per (mi,nt) ---
#pragma unroll
    for (int mi = 0; mi < 2; ++mi) {
        int mt  = mi * 8 + wave;
        int ch0 = mt * 16 + hi8;
        float bias[8];
        *(float4*)(bias)     = *(const float4*)(b1 + ch0);
        *(float4*)(bias + 4) = *(const float4*)(b1 + ch0 + 4);
#pragma unroll
        for (int nt = 0; nt < 4; ++nt) {
            union { uint4 u; __bf16 h[8]; } tmp;
#pragma unroll
            for (int r = 0; r < 8; ++r)
                tmp.h[r] = (__bf16)fmaxf(acc[mi][nt][r] + bias[r], 0.0f);
            *(uint4*)(s_y1 + mt * 1024 + (nt*16 + nl) * 16 + hi8) = tmp.u;
        }
    }
    __syncthreads();

    // --- layer 2: Y2 = relu(W2 * Y1 + b2), K = 256 ---
    f32x8 acc2[2][4] = {};
#pragma unroll
    for (int kc = 0; kc < OUT_DIMC / 32; ++kc) {
        bf16x16 a0 = load_frag32B(w2f + (((wave    ) * 8 + kc) * 32 + lane) * 16);
        bf16x16 a1 = load_frag32B(w2f + (((wave + 8) * 8 + kc) * 32 + lane) * 16);
#pragma unroll
        for (int nt = 0; nt < 4; ++nt) {
            bf16x16 bb = load_frag32B(s_y1 + (kc*2 + hlf) * 1024 + (nt*16 + nl) * 16);
            acc2[0][nt] = __builtin_amdgcn_wmma_f32_16x16x32_bf16(
                false, a0, false, bb, (short)0, acc2[0][nt], false, false);
            acc2[1][nt] = __builtin_amdgcn_wmma_f32_16x16x32_bf16(
                false, a1, false, bb, (short)0, acc2[1][nt], false, false);
        }
    }

    // --- epilogue: transpose through LDS, coalesced float4 (B,N,C) stores ---
    for (int half = 0; half < 2; ++half) {
        __syncthreads();   // s_y1 reads (or previous half copy) complete
#pragma unroll
        for (int mi = 0; mi < 2; ++mi) {
            int mt  = mi * 8 + wave;
            int ch0 = mt * 16 + hi8;
            float bias[8];
            *(float4*)(bias)     = *(const float4*)(b2 + ch0);
            *(float4*)(bias + 4) = *(const float4*)(b2 + ch0 + 4);
#pragma unroll
            for (int nt2 = 0; nt2 < 2; ++nt2) {
                int nt = half * 2 + nt2;
                float vv[8];
#pragma unroll
                for (int r = 0; r < 8; ++r)
                    vv[r] = fmaxf(acc2[mi][nt][r] + bias[r], 0.0f);
                float* dst = s_o + (nt2*16 + nl) * OUT_DIMC + ch0;
                *(float4*)(dst)     = *(const float4*)(vv);
                *(float4*)(dst + 4) = *(const float4*)(vv + 4);
            }
        }
        __syncthreads();
        for (int idx = threadIdx.x; idx < 32 * OUT_DIMC / 4; idx += 256) {
            int p  = idx >> 6;                 // point within half-tile
            int c4 = idx & 63;                 // float4 index within channels
            float4 vq = ((const float4*)s_o)[idx];
            ((float4*)(y2 + ((size_t)bi * NPTS + p0 + half*32 + p) * OUT_DIMC))[c4] = vq;
        }
    }
}

// ---------------------------------------------------------------------------
// Kernel 3: ball query (first-K-in-radius, ascending index) + gather + max.
// ---------------------------------------------------------------------------
__global__ __launch_bounds__(256) void group_kernel(const float* __restrict__ xyz,
                                                    const float* __restrict__ y2,
                                                    const float* __restrict__ child_xyz,
                                                    float* __restrict__ child_feats)
{
    const int gm = blockIdx.x;           // b*MPTS + m
    const int b  = gm / MPTS;
    const int m  = gm - b * MPTS;
    const float* xb = xyz + (size_t)b * NPTS * 3;

    __shared__ int s_idx[KNBR];

    const float cx = child_xyz[(size_t)gm*3+0];
    const float cy = child_xyz[(size_t)gm*3+1];
    const float cz = child_xyz[(size_t)gm*3+2];

    if (threadIdx.x < 32) {
        const int lane = threadIdx.x;
        int cnt = 0;
        for (int base = 0; base < NPTS && cnt < KNBR; base += 32) {
            int n = base + lane;
            float dx = xb[n*3+0]-cx, dy = xb[n*3+1]-cy, dz = xb[n*3+2]-cz;
            bool inb = (dx*dx + dy*dy + dz*dz) < (RADIUS_F * RADIUS_F);
            unsigned mask = (unsigned)__ballot(inb);
            while (mask && cnt < KNBR) {        // uniform across the wave
                int bit = __builtin_ctz(mask);
                if (lane == 0) s_idx[cnt] = base + bit;
                ++cnt;
                mask &= mask - 1;
            }
        }
        if (lane == 0) {
            int first = (cnt > 0) ? s_idx[0] : 0;
            for (int k = cnt; k < KNBR; ++k) s_idx[k] = first;
        }
    }
    __syncthreads();

    const int c = threadIdx.x;           // channel
    float mx = -3.0e38f;
#pragma unroll
    for (int k = 0; k < KNBR; ++k) {
        float v = y2[((size_t)b * NPTS + s_idx[k]) * OUT_DIMC + c];
        mx = fmaxf(mx, v);
    }
    child_feats[((size_t)b * OUT_DIMC + c) * MPTS + m] = mx;
}

// ---------------------------------------------------------------------------
extern "C" void kernel_launch(void* const* d_in, const int* in_sizes, int n_in,
                              void* d_out, int out_size, void* d_ws, size_t ws_size,
                              hipStream_t stream)
{
    const float* xyz   = (const float*)d_in[0];   // (B, N, 3)
    const float* feats = (const float*)d_in[1];   // (B, 128, N)
    const float* w1    = (const float*)d_in[2];   // (256, 128)
    const float* b1    = (const float*)d_in[3];   // (256)
    const float* w2    = (const float*)d_in[4];   // (256, 256)
    const float* b2    = (const float*)d_in[5];   // (256)
    (void)in_sizes; (void)n_in; (void)out_size; (void)ws_size;

    float* out         = (float*)d_out;
    float* child_xyz   = out;                                   // (B, M, 3)
    float* child_feats = out + (size_t)BATCH * MPTS * 3;        // (B, 256, M)

    // workspace: child_idx | w1 packed bf16 | w2 packed bf16 | y2 (B,N,256) f32
    char* ws = (char*)d_ws;
    int*    child_idx = (int*)ws;                               //  32 KB
    __bf16* w1f = (__bf16*)(ws + 32768);                        //  64 KB (32768 el)
    __bf16* w2f = (__bf16*)(ws + 32768 + 65536);                // 128 KB (65536 el)
    float*  y2  = (float*) (ws + 32768 + 65536 + 131072);       // ~33.5 MB

    fps_kernel   <<<BATCH,                 1024, 0, stream>>>(xyz, child_xyz, child_idx);
    pack_w_kernel<<<32768 / 256,            256, 0, stream>>>(w1, w1f, IN_DIMC);
    pack_w_kernel<<<65536 / 256,            256, 0, stream>>>(w2, w2f, OUT_DIMC);
    mlp_kernel   <<<BATCH * (NPTS/TILE_N),  256, 0, stream>>>(feats, w1f, b1, w2f, b2, y2);
    group_kernel <<<BATCH * MPTS,           256, 0, stream>>>(xyz, y2, child_xyz, child_feats);
}
